// MultiHeadAttention_68788196212992
// MI455X (gfx1250) — matrix operand order repaired and checked by reading, compiled, not run on hardware
//
#include <hip/hip_runtime.h>
#include <math.h>

// ---------------------------------------------------------------------------
// MI455X (gfx1250) causal multi-head attention, bf16 WMMA pipeline.
//   x[B,T,C] --QKV gemm--> qkv(bf16) --flash attn (WMMA)--> attn(bf16)
//   --proj gemm--> out[B,T,C] (f32)
// Wave32; v_wmma_f32_16x16x32_bf16; f32 accumulation; LDS-based softmax
// reductions.  Register budget kept ~110-130 VGPRs per wave (no spills):
// latency hiding comes from 8 waves/block occupancy + hoisting the V-fragment
// loads above the softmax VALU stretch.
// ---------------------------------------------------------------------------

typedef __bf16 bf16_t;
typedef __attribute__((ext_vector_type(16))) __bf16 v16bf;
typedef __attribute__((ext_vector_type(8)))  __bf16 v8bf;   // 16 bytes
typedef __attribute__((ext_vector_type(8)))  float  v8f;
typedef __attribute__((ext_vector_type(4)))  float  v4f;

#define B_  4
#define T_  2048
#define C_  1024
#define H_  16
#define D_  64
#define M_  (B_ * T_)      // 8192 rows
#define N3C (3 * C_)       // 3072

// round-to-nearest-even f32 -> bf16
__device__ __forceinline__ bf16_t f2bf(float f) {
  unsigned u = __builtin_bit_cast(unsigned, f);
  unsigned r = (u + 0x7FFFu + ((u >> 16) & 1u)) >> 16;
  return __builtin_bit_cast(bf16_t, (unsigned short)r);
}

__device__ __forceinline__ v16bf pack16(v8bf lo, v8bf hi) {
  v16bf r;
#pragma unroll
  for (int i = 0; i < 8; ++i) { r[i] = lo[i]; r[i + 8] = hi[i]; }
  return r;
}

// 32 contiguous bf16 -> one fragment (two b128 loads into adjacent VGPRs)
__device__ __forceinline__ v16bf ld32(const bf16_t* p) {
  return pack16(*reinterpret_cast<const v8bf*>(p),
                *reinterpret_cast<const v8bf*>(p + 8));
}

__device__ __forceinline__ v8f wmma_bf16(v16bf a, v16bf b, v8f c) {
  return __builtin_amdgcn_wmma_f32_16x16x32_bf16(
      false, a, false, b, (short)0, c, false, false);
}

// ---------------------------------------------------------------------------
__global__ void cvt_f32_bf16(const float* __restrict__ src,
                             bf16_t* __restrict__ dst, int n) {
  int i = blockIdx.x * blockDim.x + threadIdx.x;
  if (i < n) dst[i] = f2bf(src[i]);
}

// ---------------------------------------------------------------------------
// Out[M,N] = A[M,K](bf16) @ W[N,K]^T (bf16) + bias.  One wave -> 32x64 tile,
// two A-fragments reuse each B-fragment: 8 WMMA per 12 x 16B loads.  All six
// fragment loads of a K-step are issued as one clause before the WMMAs.
// Fragment layouts as loaded here (16-bit, wave32):
//   A: lane m=l&15; elems i = K(half*16 + i)
//   B: lane n=l&15; elems i = K(half*16 + i)  (W[N,K] row-major == B^T ready)
//   D: lane n=l&15; vgpr r = row r + half*8
// ---------------------------------------------------------------------------
template <bool OUT_F32>
__global__ void gemm32x64_bf16(const bf16_t* __restrict__ A,
                               const bf16_t* __restrict__ W,
                               const float* __restrict__ bias,
                               bf16_t* __restrict__ outb,
                               float* __restrict__ outf,
                               int M, int N, int K) {
  const int lane  = threadIdx.x & 31;
  const int wave  = threadIdx.x >> 5;
  const int tile  = blockIdx.x * (blockDim.x >> 5) + wave;
  const int tilesN = N >> 6;
  const int tm = (tile / tilesN) << 5;
  const int tn = (tile % tilesN) << 6;
  if (tm >= M) return;
  const int mr   = lane & 15;
  const int half = lane >> 4;

  const bf16_t* arow0 = A + (size_t)(tm + mr) * K;
  const bf16_t* arow1 = A + (size_t)(tm + 16 + mr) * K;
  const bf16_t* wbase = W + (size_t)(tn + mr) * K + half * 16;

  v8f acc[2][4] = {};

#pragma unroll 2
  for (int kk = 0; kk < K; kk += 32) {
    v16bf a0 =
        pack16(*reinterpret_cast<const v8bf*>(arow0 + kk + half * 16),
               *reinterpret_cast<const v8bf*>(arow0 + kk + half * 16 + 8));
    v16bf a1 =
        pack16(*reinterpret_cast<const v8bf*>(arow1 + kk + half * 16),
               *reinterpret_cast<const v8bf*>(arow1 + kk + half * 16 + 8));
    v16bf b0 = ld32(wbase + kk);
    v16bf b1 = ld32(wbase + (size_t)16 * K + kk);
    v16bf b2 = ld32(wbase + (size_t)32 * K + kk);
    v16bf b3 = ld32(wbase + (size_t)48 * K + kk);
    acc[0][0] = wmma_bf16(a0, b0, acc[0][0]);
    acc[1][0] = wmma_bf16(a1, b0, acc[1][0]);
    acc[0][1] = wmma_bf16(a0, b1, acc[0][1]);
    acc[1][1] = wmma_bf16(a1, b1, acc[1][1]);
    acc[0][2] = wmma_bf16(a0, b2, acc[0][2]);
    acc[1][2] = wmma_bf16(a1, b2, acc[1][2]);
    acc[0][3] = wmma_bf16(a0, b3, acc[0][3]);
    acc[1][3] = wmma_bf16(a1, b3, acc[1][3]);
  }

#pragma unroll
  for (int mb = 0; mb < 2; ++mb) {
#pragma unroll
    for (int nt = 0; nt < 4; ++nt) {
      const int col = tn + nt * 16 + mr;
      const float bv = bias[col];
#pragma unroll
      for (int r = 0; r < 8; ++r) {
        const int row = tm + mb * 16 + r + half * 8;
        float v = acc[mb][nt][r] + bv;
        if (OUT_F32) outf[(size_t)row * N + col] = v;
        else         outb[(size_t)row * N + col] = f2bf(v);
      }
    }
  }
}

// ---------------------------------------------------------------------------
// V^T staging: vT[b,h,d,t] = qkv[b*T+t, 2C + h*D + d]
// ---------------------------------------------------------------------------
__global__ void transpose_v(const bf16_t* __restrict__ qkv,
                            bf16_t* __restrict__ vT) {
  int i = blockIdx.x * blockDim.x + threadIdx.x;  // over B*H*D*T
  if (i >= B_ * H_ * D_ * T_) return;
  int t = i % T_;
  int d = (i / T_) % D_;
  int h = (i / (T_ * D_)) % H_;
  int b = i / (T_ * D_ * H_);
  vT[i] = qkv[(size_t)(b * T_ + t) * N3C + 2 * C_ + h * D_ + d];
}

// ---------------------------------------------------------------------------
// Flash attention, one wave per (b, h, 16-query tile); 32-key inner tiles.
// Per tile: K loads -> S WMMAs -> issue V loads -> softmax VALU stretch
// (hides V latency) -> rescale -> PV WMMAs.  Named scalar fragments only
// (no register arrays that could be demoted to scratch).
// Softmax row-ownership through LDS: scatter S (D-layout) -> sbuf[16][32],
// lane-pair (lane&15=row, lane>>4=key-half) reduces 16 contiguous keys,
// combines with partner via ONE shfl_xor(16); corr / l broadcast back to the
// D-layout lanes through tiny LDS arrays (contiguous v4f reads).
// ---------------------------------------------------------------------------
__global__ void flash_attn16(const bf16_t* __restrict__ qkv,
                             const bf16_t* __restrict__ vT,
                             bf16_t* __restrict__ outb) {
  __shared__ __align__(16) float  sbuf[8][16][32];   // 16 KB
  __shared__ __align__(16) bf16_t pbuf[8][16][32];   //  8 KB
  __shared__ __align__(16) float  corrbuf[8][16];    // .5 KB
  __shared__ __align__(16) float  lbuf[8][16];       // .5 KB

  const int lane = threadIdx.x & 31;
  const int wave = threadIdx.x >> 5;
  const int tile = blockIdx.x * (blockDim.x >> 5) + wave;
  const int qtiles = T_ / 16;
  const int qt = (tile % qtiles) * 16;
  const int h  = (tile / qtiles) % H_;
  const int b  = tile / (qtiles * H_);
  const int col  = lane & 15;
  const int half = lane >> 4;
  const int kh   = half * 16;
  const int kend = qt + 16;       // exclusive key bound (causal)
  const float scale = 0.125f;     // 1/sqrt(64)

  // Q fragments for both 32-wide D chunks, loaded once.
  const bf16_t* qrow = qkv + (size_t)(b * T_ + qt + col) * N3C + h * D_;
  v16bf aq0 = pack16(*reinterpret_cast<const v8bf*>(qrow + half * 16),
                     *reinterpret_cast<const v8bf*>(qrow + half * 16 + 8));
  v16bf aq1 = pack16(*reinterpret_cast<const v8bf*>(qrow + 32 + half * 16),
                     *reinterpret_cast<const v8bf*>(qrow + 40 + half * 16));

  v8f acc[4] = {};
  float mprev = -3.0e38f;   // row-owner state for row (lane&15)
  float lsum  = 0.0f;

  const bf16_t* kbase = qkv + (size_t)(b * T_) * N3C + C_ + h * D_ + kh;
  const bf16_t* vbase = vT + ((size_t)(b * H_ + h) * D_ + col) * T_ + kh;

  for (int kt = 0; kt < kend; kt += 32) {
    // ---- S = Q K^T : two 16x16 f32 frags (keys kt..+15 and kt+16..+31)
    v8f s0 = {}, s1 = {};
    {
      const bf16_t* kr0 = kbase + (size_t)(kt + col) * N3C;
      const bf16_t* kr1 = kbase + (size_t)(kt + 16 + col) * N3C;
      v16bf bk00 = ld32(kr0);        // keys kt..+15,  d-chunk 0
      v16bf bk10 = ld32(kr1);        // keys kt+16..31, d-chunk 0
      v16bf bk01 = ld32(kr0 + 32);   // d-chunk 1
      v16bf bk11 = ld32(kr1 + 32);
      s0 = wmma_bf16(aq0, bk00, s0);
      s1 = wmma_bf16(aq0, bk10, s1);
      s0 = wmma_bf16(aq1, bk01, s0);
      s1 = wmma_bf16(aq1, bk11, s1);
    }

    // ---- issue V fragment loads now; consumed after the softmax stretch
    v16bf bv0 = ld32(vbase + kt);
    v16bf bv1 = ld32(vbase + (size_t)16 * T_ + kt);
    v16bf bv2 = ld32(vbase + (size_t)32 * T_ + kt);
    v16bf bv3 = ld32(vbase + (size_t)48 * T_ + kt);

    // ---- scatter masked, scaled S (D-layout) into row-major LDS tile
#pragma unroll
    for (int r = 0; r < 8; ++r) {
      const int row  = r + half * 8;
      const int grow = qt + row;
      sbuf[wave][row][col]      = (kt + col <= grow)      ? s0[r] * scale : -3.0e38f;
      sbuf[wave][row][col + 16] = (kt + 16 + col <= grow) ? s1[r] * scale : -3.0e38f;
    }
    asm volatile("s_wait_dscnt 0" ::: "memory");

    // ---- row-owner softmax: lane owns row (lane&15), keys kh..kh+15
    v4f xv0 = *reinterpret_cast<const v4f*>(&sbuf[wave][col][kh]);
    v4f xv1 = *reinterpret_cast<const v4f*>(&sbuf[wave][col][kh + 4]);
    v4f xv2 = *reinterpret_cast<const v4f*>(&sbuf[wave][col][kh + 8]);
    v4f xv3 = *reinterpret_cast<const v4f*>(&sbuf[wave][col][kh + 12]);
    float m16 = xv0[0];
#pragma unroll
    for (int e = 0; e < 4; ++e) {
      m16 = fmaxf(m16, xv0[e]);
      m16 = fmaxf(m16, xv1[e]);
      m16 = fmaxf(m16, xv2[e]);
      m16 = fmaxf(m16, xv3[e]);
    }
    float tmax = fmaxf(m16, __shfl_xor(m16, 16, 32));
    float mnew = fmaxf(mprev, tmax);
    float corr = __expf(mprev - mnew);
    float psum = 0.0f;
    v8bf ph0, ph1;
#pragma unroll
    for (int e = 0; e < 4; ++e) {
      float p0 = __expf(xv0[e] - mnew);
      float p1 = __expf(xv1[e] - mnew);
      float p2 = __expf(xv2[e] - mnew);
      float p3 = __expf(xv3[e] - mnew);
      psum += p0 + p1 + p2 + p3;
      ph0[e]     = f2bf(p0);
      ph0[e + 4] = f2bf(p1);
      ph1[e]     = f2bf(p2);
      ph1[e + 4] = f2bf(p3);
    }
    psum += __shfl_xor(psum, 16, 32);
    lsum = lsum * corr + psum;
    mprev = mnew;
    *reinterpret_cast<v8bf*>(&pbuf[wave][col][kh])     = ph0;
    *reinterpret_cast<v8bf*>(&pbuf[wave][col][kh + 8]) = ph1;
    if (half == 0) corrbuf[wave][col] = corr;
    asm volatile("s_wait_dscnt 0" ::: "memory");

    // ---- rescale O accumulators (D-layout rows r+half*8, contiguous reads)
    v4f c0 = *reinterpret_cast<const v4f*>(&corrbuf[wave][half * 8]);
    v4f c1 = *reinterpret_cast<const v4f*>(&corrbuf[wave][half * 8 + 4]);
#pragma unroll
    for (int nt = 0; nt < 4; ++nt)
#pragma unroll
      for (int r = 0; r < 8; ++r)
        acc[nt][r] *= (r < 4) ? c0[r] : c1[r - 4];

    // ---- O += P V : P A-frag from LDS, V fragments already in flight
    const bf16_t* pr = &pbuf[wave][col][0];
    v16bf ap = pack16(*reinterpret_cast<const v8bf*>(pr + half * 16),
                      *reinterpret_cast<const v8bf*>(pr + half * 16 + 8));
    acc[0] = wmma_bf16(ap, bv0, acc[0]);
    acc[1] = wmma_bf16(ap, bv1, acc[1]);
    acc[2] = wmma_bf16(ap, bv2, acc[2]);
    acc[3] = wmma_bf16(ap, bv3, acc[3]);
    asm volatile("s_wait_dscnt 0" ::: "memory");
  }

  // ---- broadcast row sums to D-layout lanes, normalize, write out
  if (half == 0) lbuf[wave][col] = lsum;
  asm volatile("s_wait_dscnt 0" ::: "memory");
  v4f l0 = *reinterpret_cast<const v4f*>(&lbuf[wave][half * 8]);
  v4f l1 = *reinterpret_cast<const v4f*>(&lbuf[wave][half * 8 + 4]);
#pragma unroll
  for (int r = 0; r < 8; ++r) {
    const float inv = 1.0f / ((r < 4) ? l0[r] : l1[r - 4]);
    const size_t row = (size_t)(b * T_ + qt + r + half * 8);
#pragma unroll
    for (int nt = 0; nt < 4; ++nt)
      outb[row * C_ + h * D_ + nt * 16 + col] = f2bf(acc[nt][r] * inv);
  }
}

// ---------------------------------------------------------------------------
extern "C" void kernel_launch(void* const* d_in, const int* in_sizes, int n_in,
                              void* d_out, int out_size, void* d_ws,
                              size_t ws_size, hipStream_t stream) {
  const float* x      = (const float*)d_in[0];  // [B,T,C]
  const float* qkv_w  = (const float*)d_in[1];  // [3C,C]
  const float* qkv_b  = (const float*)d_in[2];  // [3C]
  const float* proj_w = (const float*)d_in[3];  // [C,C]
  const float* proj_b = (const float*)d_in[4];  // [C]
  float* out = (float*)d_out;                   // [B,T,C] f32

  // workspace carve-up (~109 MB total, all bf16)
  bf16_t* p = (bf16_t*)d_ws;
  bf16_t* xb     = p; p += (size_t)M_ * C_;
  bf16_t* wqkvb  = p; p += (size_t)N3C * C_;
  bf16_t* wprojb = p; p += (size_t)C_ * C_;
  bf16_t* qkvb   = p; p += (size_t)M_ * N3C;
  bf16_t* vTb    = p; p += (size_t)B_ * H_ * D_ * T_;
  bf16_t* attnb  = p; p += (size_t)M_ * C_;

  const int th = 256;
  cvt_f32_bf16<<<(M_ * C_ + th - 1) / th, th, 0, stream>>>(x, xb, M_ * C_);
  cvt_f32_bf16<<<(N3C * C_ + th - 1) / th, th, 0, stream>>>(qkv_w, wqkvb, N3C * C_);
  cvt_f32_bf16<<<(C_ * C_ + th - 1) / th, th, 0, stream>>>(proj_w, wprojb, C_ * C_);

  // QKV gemm: (8192/32)*(3072/64) = 12288 wave-tiles, 8 waves/block
  gemm32x64_bf16<false><<<12288 / 8, 256, 0, stream>>>(
      xb, wqkvb, qkv_b, qkvb, nullptr, M_, N3C, C_);

  transpose_v<<<(B_ * H_ * D_ * T_ + th - 1) / th, th, 0, stream>>>(qkvb, vTb);

  // flash attention: B*H*(T/16) = 8192 wave-tiles
  flash_attn16<<<(B_ * H_ * (T_ / 16)) / 8, 256, 0, stream>>>(qkvb, vTb, attnb);

  // projection gemm: (8192/32)*(1024/64) = 4096 wave-tiles -> f32 out
  gemm32x64_bf16<true><<<4096 / 8, 256, 0, stream>>>(
      attnb, wprojb, proj_b, nullptr, out, M_, C_, C_);
}
